// ContextEncoding_32856499814427
// MI455X (gfx1250) — matrix operand order repaired and checked by reading, not compiled, not run on hardware
//
#include <hip/hip_runtime.h>
#include <math.h>

#define Bx 32
#define Cc 256
#define Nn 4096
#define Kk 32

typedef _Float16 v16h __attribute__((ext_vector_type(16)));
typedef float    v8f  __attribute__((ext_vector_type(8)));

// ---------------------------------------------------------------- zero a_sum
__global__ void ce_zero_kernel(float* __restrict__ asum) {
    int i = blockIdx.x * blockDim.x + threadIdx.x;
    if (i < Bx * Kk) asum[i] = 0.0f;
}

// -------------------------------------------------- xsum[b,c] = sum_n x[b,c,n]
// one wave per (b,c) row; 4096 floats = 1024 float4, lane-strided.
__global__ __launch_bounds__(256) void ce_xsum_kernel(const float* __restrict__ x,
                                                      float* __restrict__ xsum) {
    int wave = (blockIdx.x * blockDim.x + threadIdx.x) >> 5;   // row id in [0, B*C)
    int lane = threadIdx.x & 31;
    const float4* row = (const float4*)(x + (size_t)wave * Nn);
    float s = 0.0f;
    #pragma unroll 4
    for (int i = lane; i < Nn / 4; i += 32) {
        float4 v = row[i];
        s += (v.x + v.y) + (v.z + v.w);
    }
    #pragma unroll
    for (int m = 16; m >= 1; m >>= 1) s += __shfl_xor(s, m, 32);
    if (lane == 0) xsum[wave] = s;
}

// ---------------- distances via WMMA + softmax + a_sum[b,k] accumulation ----
// block: 256 threads = 8 waves; each wave: 32 pixels (2 tiles of 16)
// grid.x = B * (N / 256)
__global__ __launch_bounds__(256) void ce_assign_kernel(const float* __restrict__ x,
                                                        const float* __restrict__ cw,
                                                        const float* __restrict__ scale,
                                                        float* __restrict__ asum_g) {
    __shared__ float lds_asum[Kk];
    const int tid  = threadIdx.x;
    const int wave = tid >> 5;
    const int lane = tid & 31;
    const int l15  = lane & 15;
    const int hi   = lane >> 4;          // which 16-lane half

    if (tid < Kk) lds_asum[tid] = 0.0f;
    __syncthreads();

    const int b      = blockIdx.x >> 4;          // 16 blocks per batch
    const int blk    = blockIdx.x & 15;
    const int n_base = blk * 256 + wave * 32;    // 32 pixels per wave

    // per-lane codeword constants: this lane handles k = l15 (lo) and l15+16 (hi)
    float c2lo = 0.0f, c2hi = 0.0f;
    {
        const float4* r0 = (const float4*)(cw + (size_t)l15 * Cc);
        const float4* r1 = (const float4*)(cw + (size_t)(l15 + 16) * Cc);
        #pragma unroll 4
        for (int i = 0; i < Cc / 4; ++i) {
            float4 a = r0[i], q = r1[i];
            c2lo += a.x * a.x + a.y * a.y + a.z * a.z + a.w * a.w;
            c2hi += q.x * q.x + q.y * q.y + q.z * q.z + q.w * q.w;
        }
    }
    const float slo = scale[l15];
    const float shi = scale[l15 + 16];

    v8f acc[2][2] = {};                 // [pixel tile][codeword half], fp32 C/D
    float f2p[2]  = {0.0f, 0.0f};       // per-lane partial ||f||^2

    const float* xb = x + (size_t)b * Cc * Nn;

    for (int c0 = 0; c0 < Cc; c0 += 32) {
        // ---- B tiles: 32 channels x 16 codewords, f16.
        // 16-bit B layout: lanes 0-15: elems 0..7 -> K=0..7, elems 8..15 -> K=16..23;
        // lanes 16-31: +8 on each K range; column = lane&15.
        v16h bmat[2];
        #pragma unroll
        for (int kb = 0; kb < 2; ++kb) {
            const float* src = cw + (size_t)(kb * 16 + l15) * Cc + c0 + hi * 8;
            #pragma unroll
            for (int e = 0; e < 16; e += 4) {
                float4 q = *(const float4*)(src + e + (e & 8));
                bmat[kb][e + 0] = (_Float16)q.x;
                bmat[kb][e + 1] = (_Float16)q.y;
                bmat[kb][e + 2] = (_Float16)q.z;
                bmat[kb][e + 3] = (_Float16)q.w;
            }
        }
        // ---- A tiles: 16 pixels x 32 channels, f16.
        // 16-bit A layout: lanes 0-15: elems 0..7 -> K=0..7, elems 8..15 -> K=16..23;
        // lanes 16-31: +8 on each K range. Pixel = lane&15.
        #pragma unroll
        for (int t = 0; t < 2; ++t) {
            const int n0 = n_base + t * 16 + l15;
            v16h amat;
            float f2 = 0.0f;
            #pragma unroll
            for (int e = 0; e < 16; ++e) {
                const int ch = c0 + e + ((e & 8) ? 8 : 0) + hi * 8;
                float v = xb[(size_t)ch * Nn + n0];
                amat[e] = (_Float16)v;
                f2 += v * v;
            }
            f2p[t] += f2;
            acc[t][0] = __builtin_amdgcn_wmma_f32_16x16x32_f16(
                false, amat, false, bmat[0], (short)0, acc[t][0], false, false);
            acc[t][1] = __builtin_amdgcn_wmma_f32_16x16x32_f16(
                false, amat, false, bmat[1], (short)0, acc[t][1], false, false);
        }
    }

    // ---- dist -> softmax -> a_sum.
    // C/D layout: lane&15 = codeword column; VGPR r = pixel r (+8 for upper half).
    float asum_lo = 0.0f, asum_hi = 0.0f;
    #pragma unroll
    for (int t = 0; t < 2; ++t) {
        float f2full = f2p[t] + __shfl_xor(f2p[t], 16, 32);   // both halves of pixel
        #pragma unroll
        for (int r = 0; r < 8; ++r) {
            float f2  = __shfl(f2full, r + hi * 8, 32);       // f2 of pixel this row holds
            float dlo = sqrtf(fmaxf(f2 + c2lo - 2.0f * acc[t][0][r], 0.0f));
            float dhi = sqrtf(fmaxf(f2 + c2hi - 2.0f * acc[t][1][r], 0.0f));
            float vlo = -dlo * slo;
            float vhi = -dhi * shi;
            float m = fmaxf(vlo, vhi);
            #pragma unroll
            for (int s = 1; s < 16; s <<= 1) m = fmaxf(m, __shfl_xor(m, s, 32));
            float elo = __expf(vlo - m);
            float ehi = __expf(vhi - m);
            float sum = elo + ehi;
            #pragma unroll
            for (int s = 1; s < 16; s <<= 1) sum += __shfl_xor(sum, s, 32);
            float inv = 1.0f / sum;
            asum_lo += elo * inv;
            asum_hi += ehi * inv;
        }
    }
    // combine the two halves (same k, disjoint pixel octets)
    asum_lo += __shfl_xor(asum_lo, 16, 32);
    asum_hi += __shfl_xor(asum_hi, 16, 32);
    if (lane < 16) {
        atomicAdd(&lds_asum[lane],      asum_lo);
        atomicAdd(&lds_asum[lane + 16], asum_hi);
    }
    __syncthreads();
    if (tid < Kk) atomicAdd(&asum_g[b * Kk + tid], lds_asum[tid]);
}

// ------- out[b,c] = (xsum[b,c] - sum_k asum[b,k]*cw[k,c]) / K ---------------
__global__ __launch_bounds__(256) void ce_finalize_kernel(const float* __restrict__ xsum,
                                                          const float* __restrict__ asum,
                                                          const float* __restrict__ cw,
                                                          float* __restrict__ out) {
    __shared__ float a[Kk];
    const int b = blockIdx.x;
    const int c = threadIdx.x;
    if (c < Kk) a[c] = asum[b * Kk + c];
    __syncthreads();
    float acc = xsum[b * Cc + c];
    #pragma unroll 8
    for (int k = 0; k < Kk; ++k) acc -= a[k] * cw[(size_t)k * Cc + c];
    out[b * Cc + c] = acc * (1.0f / (float)Kk);
}

// ---------------------------------------------------------------------------
extern "C" void kernel_launch(void* const* d_in, const int* in_sizes, int n_in,
                              void* d_out, int out_size, void* d_ws, size_t ws_size,
                              hipStream_t stream) {
    const float* x     = (const float*)d_in[0];   // [B,C,H,W] f32
    const float* cw    = (const float*)d_in[1];   // [K,C]    f32
    const float* scale = (const float*)d_in[2];   // [K]      f32
    float* xsum = (float*)d_ws;                   // B*C floats
    float* asum = xsum + Bx * Cc;                 // B*K floats
    float* out  = (float*)d_out;                  // [B,C]

    hipLaunchKernelGGL(ce_zero_kernel, dim3((Bx * Kk + 255) / 256), dim3(256), 0, stream, asum);
    hipLaunchKernelGGL(ce_xsum_kernel, dim3(Bx * Cc / 8), dim3(256), 0, stream, x, xsum);
    hipLaunchKernelGGL(ce_assign_kernel, dim3(Bx * (Nn / 256)), dim3(256), 0, stream,
                       x, cw, scale, asum);
    hipLaunchKernelGGL(ce_finalize_kernel, dim3(Bx), dim3(Cc), 0, stream, xsum, asum, cw, out);
}
